// DependencyGraphSheafLaplacian_75204877353146
// MI455X (gfx1250) — matrix operand order repaired and checked by reading, not compiled, not run on hardware
//
#include <hip/hip_runtime.h>
#include <math.h>
#include <stdint.h>

typedef __attribute__((ext_vector_type(2)))  float    v2f;
typedef __attribute__((ext_vector_type(8)))  float    v8f;
typedef __attribute__((ext_vector_type(16))) _Float16 v16h;

#define N_NODES 512
#define M_EDGES 4096
#define D_IN    256
#define S_DIM   8
#define NS      (N_NODES * S_DIM)   // 4096 = eigenproblem size

// ---------------------------------------------------------------------------
// Kernel 1: X = E (512x256) @ W^T (256x8)  via V_WMMA_F32_16X16X4_F32.
// One wave (32 lanes) per 16-row tile; N padded 8->16 with arithmetic masking
// (no divergent branches before the WMMA: EXEC stays all-ones).
// ---------------------------------------------------------------------------
__global__ __launch_bounds__(32) void k_project(const float* __restrict__ E,
                                                const float* __restrict__ W,
                                                float* __restrict__ X) {
  const int lane = threadIdx.x;      // 0..31, one wave per block
  const int r0   = blockIdx.x * 16;  // row tile
  const int half = lane >> 4;        // 0: lanes 0-15, 1: lanes 16-31
  const int l16  = lane & 15;
  v8f acc = {};

  const int   arow  = r0 + l16;                         // A: M = lane%16
  const float bmask = (l16 < S_DIM) ? 1.0f : 0.0f;      // pad N 8->16 with zeros
  const int   bnc   = (l16 < S_DIM) ? l16 : 0;          // clamped (no divergence)

#if __has_builtin(__builtin_amdgcn_wmma_f32_16x16x4_f32)
  for (int k0 = 0; k0 < D_IN; k0 += 4) {
    const int ka = k0 + half * 2;                       // A/B K slice per half-wave
    v2f a, b;
    a.x = E[arow * D_IN + ka + 0];                      // A 16x4: VGPR0=K, VGPR1=K+1
    a.y = E[arow * D_IN + ka + 1];
    b.x = bmask * W[bnc * D_IN + ka + 0];               // B 4x16 = W^T slice
    b.y = bmask * W[bnc * D_IN + ka + 1];
    acc = __builtin_amdgcn_wmma_f32_16x16x4_f32(
        /*neg_a=*/false, a, /*neg_b=*/false, b,
        /*c_mod=*/(short)0, acc, /*reuse_a=*/false, /*reuse_b=*/false);
  }
#else
  // Fallback: codegen-confirmed f16 WMMA (K=32 per op); slight precision loss.
  for (int k0 = 0; k0 < D_IN; k0 += 32) {
    v16h a, b;
    for (int t = 0; t < 8; ++t) {
      a[t]     = (_Float16)E[arow * D_IN + k0 +      half * 8 + t];
      a[8 + t] = (_Float16)E[arow * D_IN + k0 + 16 + half * 8 + t];
    }
    for (int t = 0; t < 8; ++t) {
      b[t]     = (_Float16)(bmask * W[bnc * D_IN + k0 +      half * 8 + t]);
      b[8 + t] = (_Float16)(bmask * W[bnc * D_IN + k0 + 16 + half * 8 + t]);
    }
    acc = __builtin_amdgcn_wmma_f32_16x16x32_f16(false, a, false, b,
                                                 (short)0, acc, false, false);
  }
#endif
  // C/D layout: VGPR g -> (M = g + 8*half, N = lane%16)
  for (int g = 0; g < 8; ++g) {
    const int row = r0 + g + half * 8;
    if (l16 < S_DIM) X[row * S_DIM + l16] = acc[g];
  }
}

// ---------------------------------------------------------------------------
// Kernel 2: zero the Laplacian accumulator.
// ---------------------------------------------------------------------------
__global__ void k_zero(float* __restrict__ p, long long n) {
  long long i = (long long)blockIdx.x * blockDim.x + threadIdx.x;
  const long long stride = (long long)gridDim.x * blockDim.x;
  for (; i < n; i += stride) p[i] = 0.0f;
}

// ---------------------------------------------------------------------------
// Kernel 3: per-edge Householder + R = W_t @ H, scatter 8x8 blocks into L.
// Exploits sparsity of delta: (src,src)+=I, (src,dst)-=R, (dst,src)-=R^T,
// (dst,dst)+=R^T R; self-loop (src==dst): reference's .set() overwrite leaves
// only -R, so the contribution is just R^T R at (dst,dst).
// One 64-thread block per edge; R staged in LDS for the R^T R pass.
// ---------------------------------------------------------------------------
__global__ __launch_bounds__(64) void k_edges(const float* __restrict__ X,
                                              const float* __restrict__ Rw,
                                              const int* __restrict__ src,
                                              const int* __restrict__ dst,
                                              const int* __restrict__ et,
                                              float* __restrict__ L) {
  __shared__ float dh[S_DIM];
  __shared__ float H[S_DIM * S_DIM];
  __shared__ float R[S_DIM * S_DIM];
  __shared__ float sc[2];
  const int e = blockIdx.x, tid = threadIdx.x;
  const int s = src[e], d = dst[e], t = et[e];

  if (tid < S_DIM) dh[tid] = X[d * S_DIM + tid] - X[s * S_DIM + tid];
  __syncthreads();
  if (tid == 0) {
    float n2 = 0.f;
    for (int i = 0; i < S_DIM; ++i) n2 += dh[i] * dh[i];
    const float nrm = sqrtf(n2) + 1e-12f;   // matches reference d_norm
    sc[0] = fminf(nrm, 1.0f);               // alpha = clamp(||d||, max=1)
    sc[1] = 1.0f / nrm;                     // for d_hat
  }
  __syncthreads();
  const float alpha = sc[0], inv = sc[1];
  const int i = tid >> 3, j = tid & 7;
  H[tid] = ((i == j) ? 1.0f : 0.0f) - alpha * (dh[i] * inv) * (dh[j] * inv);
  __syncthreads();

  const float* Wt = Rw + t * S_DIM * S_DIM;
  float r = 0.f;
  for (int k = 0; k < S_DIM; ++k) r = fmaf(Wt[i * S_DIM + k], H[k * S_DIM + j], r);
  R[tid] = r;
  __syncthreads();

  float rtr = 0.f;
  for (int k = 0; k < S_DIM; ++k) rtr = fmaf(R[k * S_DIM + i], R[k * S_DIM + j], rtr);

  atomicAdd(&L[(d * S_DIM + i) * NS + d * S_DIM + j], rtr);      // (dst,dst) += R^T R
  if (s != d) {
    if (tid < S_DIM)
      atomicAdd(&L[(s * S_DIM + tid) * NS + s * S_DIM + tid], 1.0f);  // (src,src)+=I
    atomicAdd(&L[(s * S_DIM + i) * NS + d * S_DIM + j], -r);                 // -R
    atomicAdd(&L[(d * S_DIM + i) * NS + s * S_DIM + j], -R[j * S_DIM + i]);  // -R^T
  }
}

// ---------------------------------------------------------------------------
// Householder tridiagonalization, step k (A is 4096x4096, L2-resident).
// ---------------------------------------------------------------------------
__global__ __launch_bounds__(256) void k_house(float* __restrict__ A,
                                               float* __restrict__ u,
                                               float* __restrict__ dd,
                                               float* __restrict__ ee,
                                               float* __restrict__ scal,
                                               int k, int n) {
  __shared__ float red[256];
  __shared__ float sh_alpha;
  const int tid = threadIdx.x;
  float p = 0.f;
  for (int i = k + 1 + tid; i < n; i += 256) { const float v = A[i * n + k]; p = fmaf(v, v, p); }
  red[tid] = p;
  __syncthreads();
  for (int off = 128; off > 0; off >>= 1) { if (tid < off) red[tid] += red[tid + off]; __syncthreads(); }
  if (tid == 0) {
    const float sigma = red[0];
    const float x0 = A[(k + 1) * n + k];
    const float nrm = sqrtf(sigma);
    const float alpha = (x0 >= 0.f) ? -nrm : nrm;
    dd[k] = A[k * n + k];
    ee[k] = alpha;                                  // sign irrelevant (only e^2 used)
    const float vtv = sigma - 2.f * alpha * x0 + alpha * alpha;
    scal[0] = (vtv > 1e-30f) ? 2.f / vtv : 0.f;     // beta
    sh_alpha = alpha;
  }
  __syncthreads();
  const float alpha = sh_alpha;
  for (int i = k + 1 + tid; i < n; i += 256) {
    const float v = A[i * n + k];
    u[i] = (i == k + 1) ? (v - alpha) : v;
  }
}

__global__ __launch_bounds__(256) void k_gemv(const float* __restrict__ A,
                                              const float* __restrict__ u,
                                              float* __restrict__ w, int k, int n) {
  __shared__ float red[256];
  const int row = k + 1 + blockIdx.x, tid = threadIdx.x;
  const float* __restrict__ Ar = A + row * n;
  float p = 0.f;
  for (int j = k + 1 + tid; j < n; j += 256) {
    __builtin_prefetch(Ar + j + 256, 0, 0);   // -> global_prefetch_b8 (WGP$ stream)
    p = fmaf(Ar[j], u[j], p);
  }
  red[tid] = p;
  __syncthreads();
  for (int off = 128; off > 0; off >>= 1) { if (tid < off) red[tid] += red[tid + off]; __syncthreads(); }
  if (tid == 0) w[row] = red[0];
}

__global__ __launch_bounds__(256) void k_wprime(const float* __restrict__ u,
                                                const float* __restrict__ w,
                                                float* __restrict__ w2,
                                                const float* __restrict__ scal,
                                                int k, int n) {
  __shared__ float red[256];
  const int tid = threadIdx.x;
  float p = 0.f;
  for (int i = k + 1 + tid; i < n; i += 256) p = fmaf(u[i], w[i], p);
  red[tid] = p;
  __syncthreads();
  for (int off = 128; off > 0; off >>= 1) { if (tid < off) red[tid] += red[tid + off]; __syncthreads(); }
  const float K = red[0];
  const float beta = scal[0];
  const float c = 0.5f * beta * beta * K;
  for (int i = k + 1 + tid; i < n; i += 256) w2[i] = beta * w[i] - c * u[i];
}

// Rank-2 update over the trailing (m x m) submatrix: 2D grid, no integer
// division (this loop body executes n^3/3 times across the whole solve; it
// must be pure load-fma-fma-store against L2). Row scalars u[i], w2[i] are
// block-uniform -> scalar loads; column stream is coalesced.
__global__ __launch_bounds__(256) void k_rank2(float* __restrict__ A,
                                               const float* __restrict__ u,
                                               const float* __restrict__ w2,
                                               int k, int n) {
  const int i  = k + 1 + blockIdx.y;                       // row (uniform per block)
  const int j  = k + 1 + blockIdx.x * 256 + threadIdx.x;   // column
  if (j >= n) return;
  const float ui = u[i], wi = w2[i];
  float* __restrict__ Ar = A + i * n;
  Ar[j] -= ui * w2[j] + wi * u[j];
}

__global__ void k_finalize(const float* __restrict__ A, float* __restrict__ dd,
                           float* __restrict__ ee, int n) {
  dd[n - 2] = A[(n - 2) * n + (n - 2)];
  dd[n - 1] = A[(n - 1) * n + (n - 1)];
  ee[n - 2] = A[(n - 1) * n + (n - 2)];
}

// ---------------------------------------------------------------------------
// Sturm-sequence bisection: one thread per eigenvalue (ascending by index).
// d and e^2 staged in LDS (each thread re-scans all 4096 entries ~48x).
// Diagonal staged via async-to-LDS DMA (ASYNCcnt path).
// ---------------------------------------------------------------------------
__global__ __launch_bounds__(256) void k_bisect(const float* __restrict__ dd,
                                                const float* __restrict__ ee,
                                                float* __restrict__ out, int n) {
  __shared__ float sd[NS];
  __shared__ float se2[NS];
  __shared__ float sea[NS];
  const int tid = threadIdx.x;

  const unsigned long long dbase = (unsigned long long)(uintptr_t)dd;
  for (int i = tid; i < n; i += 256) {
    const uint32_t lds_off = (uint32_t)(uintptr_t)&sd[i];
    asm volatile("global_load_async_to_lds_b32 %0, %1, %2 offset:0"
                 :: "v"(lds_off), "v"(i * 4), "s"(dbase) : "memory");
  }
  for (int i = tid; i < n; i += 256) {
    const float e = (i < n - 1) ? ee[i] : 0.f;
    se2[i] = e * e;
    sea[i] = fabsf(e);
  }
  asm volatile("s_wait_asynccnt 0" ::: "memory");
  __syncthreads();

  // Gershgorin bounds
  float lo = 1e30f, hi = -1e30f;
  for (int i = 0; i < n; ++i) {
    const float el = (i > 0) ? sea[i - 1] : 0.f;
    const float r = el + sea[i];
    lo = fminf(lo, sd[i] - r);
    hi = fmaxf(hi, sd[i] + r);
  }

  const int idx = blockIdx.x * 256 + tid;  // 0..n-1, ascending
  float xlo = lo, xhi = hi;
  for (int it = 0; it < 48; ++it) {
    const float mid = 0.5f * (xlo + xhi);
    int cnt = 0;
    float q = sd[0] - mid;
    cnt += (q < 0.f);
    for (int i = 1; i < n; ++i) {
      const float qa = fabsf(q);
      const float den = (qa < 1e-20f) ? ((q < 0.f) ? -1e-20f : 1e-20f) : q;
      q = sd[i] - mid - se2[i - 1] / den;
      cnt += (q < 0.f);
    }
    if (cnt > idx) xhi = mid; else xlo = mid;
  }
  out[idx] = fmaxf(0.5f * (xlo + xhi), 0.0f);
}

// ---------------------------------------------------------------------------
extern "C" void kernel_launch(void* const* d_in, const int* in_sizes, int n_in,
                              void* d_out, int out_size, void* d_ws, size_t ws_size,
                              hipStream_t stream) {
  (void)in_sizes; (void)n_in; (void)out_size; (void)ws_size;
  const float* E  = (const float*)d_in[0];   // node_embeddings (512,256)
  const float* W  = (const float*)d_in[1];   // proj_W (8,256)
  const float* Rw = (const float*)d_in[2];   // R_weights (8,8,8)
  const int*   sr = (const int*)d_in[3];
  const int*   ds = (const int*)d_in[4];
  const int*   et = (const int*)d_in[5];
  float* out = (float*)d_out;

  const int n = NS;
  float* A    = (float*)d_ws;                 // 4096x4096 Laplacian (64 MB, L2-resident)
  float* X    = A + (size_t)n * n;            // 512x8 projected features
  float* u    = X + NS;
  float* w    = u + n;
  float* w2   = w + n;
  float* dd   = w2 + n;
  float* ee   = dd + n;
  float* scal = ee + n;

  k_project<<<N_NODES / 16, 32, 0, stream>>>(E, W, X);
  k_zero<<<4096, 256, 0, stream>>>(A, (long long)n * n);
  k_edges<<<M_EDGES, 64, 0, stream>>>(X, Rw, sr, ds, et, A);

  for (int k = 0; k <= n - 3; ++k) {
    const int m = n - k - 1;
    k_house<<<1, 256, 0, stream>>>(A, u, dd, ee, scal, k, n);
    k_gemv<<<m, 256, 0, stream>>>(A, u, w, k, n);
    k_wprime<<<1, 256, 0, stream>>>(u, w, w2, scal, k, n);
    dim3 g2((unsigned)((m + 255) / 256), (unsigned)m, 1);
    k_rank2<<<g2, 256, 0, stream>>>(A, u, w2, k, n);
  }
  k_finalize<<<1, 1, 0, stream>>>(A, dd, ee, n);
  k_bisect<<<n / 256, 256, 0, stream>>>(dd, ee, out, n);
}
